// MPNNModel_83004537962838
// MI455X (gfx1250) — compile-verified, matrix operand order checked by reading, not run on hardware
//
#include <hip/hip_runtime.h>

// ---------------------------------------------------------------------------
// MPNN forward for MI455X (gfx1250). Big GEMMs (edge MLP, fused GRU) run on
// v_wmma_f32_16x16x32_bf16. Weights are pre-swizzled into WMMA-fragment-major
// bf16 so every B fragment is two coalesced global_load_b128 per lane; A
// fragments are two ds_load_b128 from conflict-free-strided LDS tiles. Each
// wave owns a 32x32 output tile so B fragments are loaded once per block.
// ---------------------------------------------------------------------------

typedef __bf16 bf16;
typedef bf16  v16bf __attribute__((ext_vector_type(16)));
typedef float v8f   __attribute__((ext_vector_type(8)));

#define HID  256
#define EDIM 64
#define KIN  576            // 2*HID + EDIM
#define ASTR 648            // row stride (bf16): 1296B rows, 324 dwords ≡ 4 (mod 64)
#define TSTR 264            // 528B rows, 132 dwords ≡ 4 (mod 64)
#define GSTR 520            // 1040B rows, 260 dwords ≡ 4 (mod 64)

union FragU { v16bf v; uint4 u[2]; };

__device__ __forceinline__ float siluf(float x){ return x * (1.0f/(1.0f+__expf(-x))); }
__device__ __forceinline__ float sigmf(float x){ return 1.0f/(1.0f+__expf(-x)); }
__device__ __forceinline__ unsigned fkey(float f){
  unsigned b = __float_as_uint(f);
  return b ^ ((unsigned)((int)b >> 31) | 0x80000000u);
}
__device__ __forceinline__ float funkey(unsigned k){
  unsigned b = (k & 0x80000000u) ? (k ^ 0x80000000u) : ~k;
  return __uint_as_float(b);
}

// ------------------------------ prep kernels -------------------------------
// Pack row-major f32 W[K][C] into WMMA-B-fragment-major bf16:
// out[((kt*(C/16) + nt)*32 + lane)*16 + e] = W[kt*32 + kof(e,lane)][nt*16 + lane%16]
__global__ void pack_b_kernel(const float* __restrict__ W, bf16* __restrict__ out,
                              int K, int C){
  int i = blockIdx.x * 256 + threadIdx.x;
  if (i >= K * C) return;
  int e    = i & 15;
  int lane = (i >> 4) & 31;
  int nt   = (i >> 9) % (C >> 4);
  int kt   = (i >> 9) / (C >> 4);
  int k = kt*32 + ((e < 8) ? e : (8 + e)) + 8*(lane >> 4);
  int c = nt*16 + (lane & 15);
  out[i] = (bf16)W[k*C + c];
}

// Stacked GRU weight, fragment-packed. Logical Wg[512][1024]:
// cols 0..511  : [W_ih;W_hh] (r,z gate sums),  bias b_ih+b_hh
// cols 512..767: [W_ih[:,512:768]; 0]  (inn),  bias b_ih
// cols 768..1023:[0; W_hh[:,512:768]]  (hn),   bias b_hh
__device__ __forceinline__ float gru_w(const float* Wih, const float* Whh, int k, int c){
  if (k < 256) return (c < 768) ? Wih[k*768 + c] : 0.0f;
  int kk = k - 256;
  return (c < 512) ? Whh[kk*768 + c] : ((c >= 768) ? Whh[kk*768 + (c-256)] : 0.0f);
}
__global__ void build_gru_kernel(const float* __restrict__ Wih, const float* __restrict__ Whh,
                                 const float* __restrict__ bih, const float* __restrict__ bhh,
                                 bf16* __restrict__ Wg, float* __restrict__ bg){
  int i = blockIdx.x * 256 + threadIdx.x;
  if (i < 512*1024) {
    int e    = i & 15;
    int lane = (i >> 4) & 31;
    int nt   = (i >> 9) & 63;       // 1024/16 tiles
    int kt   = (i >> 9) >> 6;
    int k = kt*32 + ((e < 8) ? e : (8 + e)) + 8*(lane >> 4);
    int c = nt*16 + (lane & 15);
    Wg[i] = (bf16)gru_w(Wih, Whh, k, c);
  }
  if (i < 1024) {
    bg[i] = (i < 512) ? (bih[i] + bhh[i]) : ((i < 768) ? bih[i] : bhh[i-256]);
  }
}

__global__ void init_gmax_kernel(unsigned* __restrict__ p, int n){
  int i = blockIdx.x * 256 + threadIdx.x;
  if (i < n) p[i] = fkey(-INFINITY);
}

// ------------------------------ atom encoder -------------------------------
__global__ __launch_bounds__(256)
void atom_encoder_kernel(const int* __restrict__ x_cat, const float* __restrict__ x_cont,
                         const float* __restrict__ e0, const float* __restrict__ e1,
                         const float* __restrict__ e2, const float* __restrict__ e3,
                         const float* __restrict__ e4,
                         const float* __restrict__ W, const float* __restrict__ b,
                         const float* __restrict__ g, const float* __restrict__ be,
                         float* __restrict__ h, bf16* __restrict__ h_bf, int N){
  __shared__ float cat[80];
  __shared__ float red[256], red2[256];
  int n = blockIdx.x, t = threadIdx.x;
  if (t < 75) {
    float v;
    if      (t < 32) v = e0[x_cat[n*5+0]*32 + t];
    else if (t < 48) v = e1[x_cat[n*5+1]*16 + (t-32)];
    else if (t < 56) v = e2[x_cat[n*5+2]*8  + (t-48)];
    else if (t < 64) v = e3[x_cat[n*5+3]*8  + (t-56)];
    else if (t < 72) v = e4[x_cat[n*5+4]*8  + (t-64)];
    else             v = x_cont[n*3 + (t-72)];
    cat[t] = v;
  }
  __syncthreads();
  float acc = b[t];
  for (int k = 0; k < 75; ++k) acc += cat[k] * W[k*HID + t];
  float y = siluf(acc);
  red[t] = y; red2[t] = y*y;
  __syncthreads();
  for (int s = 128; s > 0; s >>= 1){
    if (t < s){ red[t] += red[t+s]; red2[t] += red2[t+s]; }
    __syncthreads();
  }
  float m = red[0] * (1.0f/HID);
  float var = red2[0] * (1.0f/HID) - m*m;
  float out = (y - m) * rsqrtf(var + 1e-5f) * g[t] + be[t];
  h[(size_t)n*HID + t] = out;
  h_bf[(size_t)n*HID + t] = (bf16)out;
}

// ------------------------------ bond encoder -------------------------------
__global__ __launch_bounds__(256)
void bond_encoder_kernel(const int* __restrict__ edge_cat, const float* __restrict__ edge_cont,
                         const float* __restrict__ eb0, const float* __restrict__ eb1,
                         const float* __restrict__ W, const float* __restrict__ b,
                         const float* __restrict__ g, const float* __restrict__ be,
                         bf16* __restrict__ e_bf, int E){
  __shared__ float cat[4][32];
  __shared__ float red[4][64], red2[4][64];
  int sub = threadIdx.x >> 6, j = threadIdx.x & 63;
  int e = blockIdx.x * 4 + sub;
  bool valid = e < E;
  if (valid && j < 26) {
    float v;
    if      (j < 16) v = eb0[edge_cat[e*2+0]*16 + j];
    else if (j < 24) v = eb1[edge_cat[e*2+1]*8 + (j-16)];
    else             v = edge_cont[e*2 + (j-24)];
    cat[sub][j] = v;
  }
  __syncthreads();
  float y = 0.0f;
  if (valid) {
    float acc = b[j];
    for (int k = 0; k < 26; ++k) acc += cat[sub][k] * W[k*EDIM + j];
    y = siluf(acc);
    red[sub][j] = y; red2[sub][j] = y*y;
  }
  __syncthreads();
  for (int s = 32; s > 0; s >>= 1){
    if (valid && j < s){ red[sub][j] += red[sub][j+s]; red2[sub][j] += red2[sub][j+s]; }
    __syncthreads();
  }
  if (valid) {
    float m = red[sub][0] * (1.0f/EDIM);
    float var = red2[sub][0] * (1.0f/EDIM) - m*m;
    e_bf[(size_t)e*EDIM + j] = (bf16)((y - m) * rsqrtf(var + 1e-5f) * g[j] + be[j]);
  }
}

// -------------------- edge MLP + scatter (WMMA bf16) -----------------------
// 32 edges / block, 8 waves. Wave w owns the full 32 rows x 32 cols tile at
// cols [w*32, w*32+32): B fragments loaded once per block and reused across
// both m-tiles. A: 2x ds_load_b128; B: 2x global_load_b128 (packed weights).
__global__ __launch_bounds__(256)
void edge_mlp_kernel(const bf16* __restrict__ h_bf, const bf16* __restrict__ e_bf,
                     const int* __restrict__ edge_index,
                     const bf16* __restrict__ W1p, const float* __restrict__ b1,
                     const bf16* __restrict__ W2p, const float* __restrict__ b2,
                     float* __restrict__ aggr, int E){
  __shared__ __align__(16) bf16 sA[32 * ASTR];   // 41472 B; aliased as T[32][TSTR]
  bf16* sT = sA;

  const int tid  = threadIdx.x;
  const int wave = tid >> 5;
  const int lane = tid & 31;
  const int lg   = lane >> 4;
  const int lr   = lane & 15;
  const int e0   = blockIdx.x * 32;
  const int* srcIdx = edge_index;        // edge_index[0,:]
  const int* dstIdx = edge_index + E;    // edge_index[1,:]

  // ---- stage A = [h[dst] | h[src] | e] (bf16) ----
  {
    const int er = tid >> 3, part = tid & 7;
    const int eg = e0 + er;
    if (eg < E) {
      const int d = dstIdx[eg], s = srcIdx[eg];
      const uint4* hd = (const uint4*)(h_bf + (size_t)d  * HID);
      const uint4* hs = (const uint4*)(h_bf + (size_t)s  * HID);
      const uint4* ee = (const uint4*)(e_bf + (size_t)eg * EDIM);
      uint4* rowD = (uint4*)(sA + er * ASTR);
      uint4* rowS = (uint4*)(sA + er * ASTR + HID);
      uint4* rowE = (uint4*)(sA + er * ASTR + 2*HID);
      #pragma unroll
      for (int q = 0; q < 4; ++q) { rowD[part + q*8] = hd[part + q*8]; rowS[part + q*8] = hs[part + q*8]; }
      rowE[part] = ee[part];
    } else {
      uint4 z = make_uint4(0,0,0,0);
      uint4* row = (uint4*)(sA + er * ASTR);
      #pragma unroll
      for (int q = 0; q < 9; ++q) row[part + q*8] = z;
    }
  }
  __syncthreads();

  const int colbase = wave * 32;

  // ---- GEMM1: t = silu(A @ W1 + b1), K=576 ----
  v8f acc[2][2] = {};
  for (int kt = 0; kt < KIN/32; ++kt) {
    FragU af[2];
    #pragma unroll
    for (int mt = 0; mt < 2; ++mt) {
      const uint4* ap = (const uint4*)(sA + (mt*16 + lr) * ASTR + kt*32 + lg*8);
      af[mt].u[0] = ap[0]; af[mt].u[1] = ap[2];
    }
    #pragma unroll
    for (int ntl = 0; ntl < 2; ++ntl) {
      const int nt = wave*2 + ntl;
      const uint4* bp = (const uint4*)(W1p + ((size_t)(kt*16 + nt)*32 + lane)*16);
      FragU bfr; bfr.u[0] = bp[0]; bfr.u[1] = bp[1];
      #pragma unroll
      for (int mt = 0; mt < 2; ++mt)
        acc[mt][ntl] = __builtin_amdgcn_wmma_f32_16x16x32_bf16(false, af[mt].v, false, bfr.v,
                                                               (short)0, acc[mt][ntl], false, false);
    }
  }
  __syncthreads();   // all waves done reading sA before aliasing it as T

  // ---- epilogue 1: T = silu(acc + b1) as bf16 ----
  #pragma unroll
  for (int ntl = 0; ntl < 2; ++ntl) {
    int col = colbase + ntl*16 + lr;
    float bias = b1[col];
    #pragma unroll
    for (int mt = 0; mt < 2; ++mt) {
      #pragma unroll
      for (int i = 0; i < 8; ++i) {
        int row = mt*16 + i + 8*lg;
        sT[row * TSTR + col] = (bf16)siluf(acc[mt][ntl][i] + bias);
      }
    }
  }
  __syncthreads();

  // ---- GEMM2: msg = T @ W2 + b2, K=256 ----
  v8f acc2[2][2] = {};
  for (int kt = 0; kt < HID/32; ++kt) {
    FragU af[2];
    #pragma unroll
    for (int mt = 0; mt < 2; ++mt) {
      const uint4* ap = (const uint4*)(sT + (mt*16 + lr) * TSTR + kt*32 + lg*8);
      af[mt].u[0] = ap[0]; af[mt].u[1] = ap[2];
    }
    #pragma unroll
    for (int ntl = 0; ntl < 2; ++ntl) {
      const int nt = wave*2 + ntl;
      const uint4* bp = (const uint4*)(W2p + ((size_t)(kt*16 + nt)*32 + lane)*16);
      FragU bfr; bfr.u[0] = bp[0]; bfr.u[1] = bp[1];
      #pragma unroll
      for (int mt = 0; mt < 2; ++mt)
        acc2[mt][ntl] = __builtin_amdgcn_wmma_f32_16x16x32_bf16(false, af[mt].v, false, bfr.v,
                                                                (short)0, acc2[mt][ntl], false, false);
    }
  }

  // ---- scatter: aggr[dst] += msg ----
  int dsts[2][8];
  #pragma unroll
  for (int mt = 0; mt < 2; ++mt) {
    #pragma unroll
    for (int i = 0; i < 8; ++i) {
      int eg = e0 + mt*16 + i + 8*lg;
      dsts[mt][i] = (eg < E) ? dstIdx[eg] : -1;
    }
  }
  #pragma unroll
  for (int ntl = 0; ntl < 2; ++ntl) {
    int col = colbase + ntl*16 + lr;
    float bias = b2[col];
    #pragma unroll
    for (int mt = 0; mt < 2; ++mt) {
      #pragma unroll
      for (int i = 0; i < 8; ++i) {
        if (dsts[mt][i] >= 0)
          atomicAdd(aggr + (size_t)dsts[mt][i]*HID + col, acc2[mt][ntl][i] + bias);
      }
    }
  }
}

// ------------------------ GRU + LayerNorm (WMMA bf16) ----------------------
// 16 nodes / block, 8 waves. A = [aggr|h] [16x512] bf16 in LDS; Wg fragment-
// packed [512x1024] bf16 in global (columns fully partitioned across waves).
// Wave w owns cols {w*32+q0*16+lr} in all four gate regions so the GRU
// combine is register-local.
__global__ __launch_bounds__(256)
void gru_ln_kernel(const float* __restrict__ aggr, const float* __restrict__ h,
                   const bf16* __restrict__ h_bf,
                   const bf16* __restrict__ Wg, const float* __restrict__ bg,
                   const float* __restrict__ ln_g, const float* __restrict__ ln_b,
                   float* __restrict__ h_out, bf16* __restrict__ hbf_out,
                   int N, int addResidual){
  __shared__ __align__(16) bf16 sA[16 * GSTR];   // 16640 B
  __shared__ float sSum[16], sSq[16];

  const int tid  = threadIdx.x;
  const int wave = tid >> 5;
  const int lane = tid & 31;
  const int lg   = lane >> 4;
  const int lr   = lane & 15;
  const int node0 = blockIdx.x * 16;

  { // stage A
    int nl = tid >> 4, part = tid & 15;
    int n = node0 + nl;
    if (part < 8) {
      bf16* d = sA + nl*GSTR + part*32;
      if (n < N) {
        const float* s = aggr + (size_t)n*HID + part*32;
        #pragma unroll
        for (int q = 0; q < 32; ++q) d[q] = (bf16)s[q];
      } else {
        #pragma unroll
        for (int q = 0; q < 32; ++q) d[q] = (bf16)0.0f;
      }
    } else {
      uint4* d = (uint4*)(sA + nl*GSTR + HID + (part-8)*32);
      if (n < N) {
        const uint4* s = (const uint4*)(h_bf + (size_t)n*HID + (part-8)*32);
        #pragma unroll
        for (int q = 0; q < 4; ++q) d[q] = s[q];
      } else {
        uint4 z = make_uint4(0,0,0,0);
        #pragma unroll
        for (int q = 0; q < 4; ++q) d[q] = z;
      }
    }
  }
  if (tid < 16) { sSum[tid] = 0.0f; sSq[tid] = 0.0f; }
  __syncthreads();

  v8f acc[8] = {};
  for (int kt = 0; kt < 16; ++kt) {
    FragU af;
    const uint4* ap = (const uint4*)(sA + lr*GSTR + kt*32 + lg*8);
    af.u[0] = ap[0]; af.u[1] = ap[2];
    #pragma unroll
    for (int q = 0; q < 8; ++q) {
      const int gate = q >> 1, q0 = q & 1;
      const int nt = gate*16 + wave*2 + q0;          // col tile of 1024-wide output
      const uint4* bp = (const uint4*)(Wg + ((size_t)(kt*64 + nt)*32 + lane)*16);
      FragU bfr; bfr.u[0] = bp[0]; bfr.u[1] = bp[1];
      acc[q] = __builtin_amdgcn_wmma_f32_16x16x32_bf16(false, af.v, false, bfr.v,
                                                       (short)0, acc[q], false, false);
    }
  }

  // GRU combine (per-lane) + LN statistics
  float hv[16];
  #pragma unroll
  for (int q0 = 0; q0 < 2; ++q0) {
    const int c = wave*32 + q0*16 + lr;
    const float br = bg[c], bz = bg[c+256], bn = bg[c+512], bh = bg[c+768];
    #pragma unroll
    for (int i = 0; i < 8; ++i) {
      const int nl = i + 8*lg;
      const int n = node0 + nl;
      float r  = sigmf(acc[0+q0][i] + br);
      float z  = sigmf(acc[2+q0][i] + bz);
      float nn = acc[4+q0][i] + bn;
      float hn = acc[6+q0][i] + bh;
      float nv = tanhf(nn + r*hn);
      float hold = (n < N) ? h[(size_t)n*HID + c] : 0.0f;
      float out = (1.0f - z)*nv + z*hold + (addResidual ? hold : 0.0f);
      hv[q0*8 + i] = out;
      if (n < N) { atomicAdd(&sSum[nl], out); atomicAdd(&sSq[nl], out*out); }
    }
  }
  __syncthreads();
  #pragma unroll
  for (int q0 = 0; q0 < 2; ++q0) {
    const int c = wave*32 + q0*16 + lr;
    const float gln = ln_g[c], bln = ln_b[c];
    #pragma unroll
    for (int i = 0; i < 8; ++i) {
      const int nl = i + 8*lg;
      const int n = node0 + nl;
      if (n < N) {
        float m = sSum[nl] * (1.0f/HID);
        float var = sSq[nl] * (1.0f/HID) - m*m;
        float y = (hv[q0*8 + i] - m) * rsqrtf(var + 1e-5f) * gln + bln;
        h_out[(size_t)n*HID + c] = y;
        hbf_out[(size_t)n*HID + c] = (bf16)y;
      }
    }
  }
}

// ------------------------------- readout -----------------------------------
__global__ __launch_bounds__(128)
void gate_kernel(const float* __restrict__ h, const float* __restrict__ gW1,
                 const float* __restrict__ gb1, const float* __restrict__ gW2,
                 const float* __restrict__ gb2, const int* __restrict__ batch,
                 float* __restrict__ gates, unsigned* __restrict__ gmax, int N){
  __shared__ float hs[256]; __shared__ float red[128];
  int n = blockIdx.x, t = threadIdx.x;
  hs[t] = h[(size_t)n*HID + t]; hs[t+128] = h[(size_t)n*HID + t + 128];
  __syncthreads();
  float acc = gb1[t];
  for (int k = 0; k < 256; ++k) acc += hs[k] * gW1[k*128 + t];
  red[t] = siluf(acc) * gW2[t];
  __syncthreads();
  for (int s = 64; s > 0; s >>= 1){ if (t < s) red[t] += red[t+s]; __syncthreads(); }
  if (t == 0) {
    float gv = red[0] + gb2[0];
    gates[n] = gv;
    atomicMax(gmax + batch[n], fkey(gv));
  }
}

__global__ void expden_kernel(const float* __restrict__ gates, const unsigned* __restrict__ gmax,
                              const int* __restrict__ batch, float* __restrict__ expg,
                              float* __restrict__ den, int N){
  int i = blockIdx.x * 256 + threadIdx.x;
  if (i < N) {
    float gm = funkey(gmax[batch[i]]);
    if (!__builtin_isfinite(gm)) gm = 0.0f;
    float eg = __expf(gates[i] - gm);
    expg[i] = eg;
    atomicAdd(den + batch[i], eg);
  }
}

__global__ __launch_bounds__(256)
void value_kernel(const float* __restrict__ h, const float* __restrict__ vW,
                  const float* __restrict__ vb, const float* __restrict__ expg,
                  const float* __restrict__ den, const int* __restrict__ batch,
                  float* __restrict__ hgraph, int N){
  __shared__ float hs[256];
  int n = blockIdx.x, t = threadIdx.x;
  hs[t] = h[(size_t)n*HID + t];
  __syncthreads();
  float acc = vb[t];
  for (int k = 0; k < 256; ++k) acc += hs[k] * vW[k*HID + t];
  float v = siluf(acc);
  int bg = batch[n];
  float w = expg[n] / (den[bg] + 1e-16f);
  atomicAdd(hgraph + (size_t)bg*HID + t, w * v);
}

__global__ __launch_bounds__(128)
void head_kernel(const float* __restrict__ hgraph, const float* __restrict__ sW,
                 const float* __restrict__ sb, const float* __restrict__ sg,
                 const float* __restrict__ sbe, const float* __restrict__ hW1,
                 const float* __restrict__ hb1, const float* __restrict__ hW2,
                 const float* __restrict__ hb2, float* __restrict__ out, int G){
  __shared__ float hs[256]; __shared__ float red[128], red2[128];
  __shared__ float sh[128]; __shared__ float u[64];
  int gi = blockIdx.x, t = threadIdx.x;
  hs[t] = hgraph[(size_t)gi*HID + t]; hs[t+128] = hgraph[(size_t)gi*HID + t + 128];
  __syncthreads();
  float acc = sb[t];
  for (int k = 0; k < 256; ++k) acc += hs[k] * sW[k*128 + t];
  float y = siluf(acc);
  red[t] = y; red2[t] = y*y;
  __syncthreads();
  for (int s = 64; s > 0; s >>= 1){ if (t < s){ red[t]+=red[t+s]; red2[t]+=red2[t+s]; } __syncthreads(); }
  float m = red[0] * (1.0f/128.0f);
  float var = red2[0] * (1.0f/128.0f) - m*m;
  sh[t] = (y - m) * rsqrtf(var + 1e-5f) * sg[t] + sbe[t];
  __syncthreads();
  if (t < 64) {
    float a2 = hb1[t];
    for (int k = 0; k < 128; ++k) a2 += sh[k] * hW1[k*64 + t];
    u[t] = siluf(a2) * hW2[t];
  }
  __syncthreads();
  for (int s = 32; s > 0; s >>= 1){ if (t < s) u[t] += u[t+s]; __syncthreads(); }
  if (t == 0) out[gi] = u[0] + hb2[0];
}

// ------------------------------- launcher ----------------------------------
extern "C" void kernel_launch(void* const* d_in, const int* in_sizes, int n_in,
                              void* d_out, int out_size, void* d_ws, size_t ws_size,
                              hipStream_t stream) {
  const int* x_cat      = (const int*)  d_in[0];
  const float* x_cont   = (const float*)d_in[1];
  const int* edge_cat   = (const int*)  d_in[2];
  const float* edge_cont= (const float*)d_in[3];
  const int* edge_index = (const int*)  d_in[4];
  const int* batch      = (const int*)  d_in[5];
  const int N = in_sizes[5];
  const int E = in_sizes[3] / 2;
  const int G = out_size;

  // params flattened in insertion order
  const float* aW  = (const float*)d_in[11];
  const float* ab  = (const float*)d_in[12];
  const float* ag  = (const float*)d_in[13];
  const float* abe = (const float*)d_in[14];
  const float* bW  = (const float*)d_in[17];
  const float* bb  = (const float*)d_in[18];
  const float* bg_ = (const float*)d_in[19];
  const float* bbe = (const float*)d_in[20];
  // mp layer i: base 21 + i*10 : W1,b1,W2,b2,W_ih,b_ih,W_hh,b_hh,ln_g,ln_b
  const float* gW1 = (const float*)d_in[61];
  const float* gb1 = (const float*)d_in[62];
  const float* gW2 = (const float*)d_in[63];
  const float* gb2 = (const float*)d_in[64];
  const float* vW  = (const float*)d_in[65];
  const float* vb  = (const float*)d_in[66];
  const float* sW  = (const float*)d_in[67];
  const float* sb  = (const float*)d_in[68];
  const float* sg  = (const float*)d_in[69];
  const float* sbe = (const float*)d_in[70];
  const float* hW1 = (const float*)d_in[71];
  const float* hb1 = (const float*)d_in[72];
  const float* hW2 = (const float*)d_in[73];
  const float* hb2 = (const float*)d_in[74];

  // workspace bump allocator (256B aligned)
  char* ws = (char*)d_ws; size_t off = 0;
  auto alloc = [&](size_t bytes) -> void* {
    void* p = ws + off; off = (off + bytes + 255) & ~(size_t)255; return p;
  };
  float* h       = (float*)alloc((size_t)N * HID * 4);
  bf16*  h_bf    = (bf16*) alloc((size_t)N * HID * 2);
  bf16*  e_bf    = (bf16*) alloc((size_t)E * EDIM * 2);
  float* aggr    = (float*)alloc((size_t)N * HID * 4);
  bf16*  W1bf    = (bf16*) alloc((size_t)4 * KIN * HID * 2);
  bf16*  W2bf    = (bf16*) alloc((size_t)4 * HID * HID * 2);
  bf16*  Wgbf    = (bf16*) alloc((size_t)4 * 512 * 1024 * 2);
  float* bgru    = (float*)alloc((size_t)4 * 1024 * 4);
  float* gates   = (float*)alloc((size_t)N * 4);
  float* expg    = (float*)alloc((size_t)N * 4);
  unsigned* gmax = (unsigned*)alloc((size_t)G * 4);
  float* den     = (float*)alloc((size_t)G * 4);
  float* hgraph  = (float*)alloc((size_t)G * HID * 4);
  (void)ws_size; (void)n_in;

  // ---- prep: fragment-pack weights (bf16) ----
  for (int l = 0; l < 4; ++l) {
    const float* W1 = (const float*)d_in[21 + l*10 + 0];
    const float* W2 = (const float*)d_in[21 + l*10 + 2];
    const float* Wih= (const float*)d_in[21 + l*10 + 4];
    const float* bih= (const float*)d_in[21 + l*10 + 5];
    const float* Whh= (const float*)d_in[21 + l*10 + 6];
    const float* bhh= (const float*)d_in[21 + l*10 + 7];
    pack_b_kernel<<<(KIN*HID + 255)/256, 256, 0, stream>>>(W1, W1bf + (size_t)l*KIN*HID, KIN, HID);
    pack_b_kernel<<<(HID*HID + 255)/256, 256, 0, stream>>>(W2, W2bf + (size_t)l*HID*HID, HID, HID);
    build_gru_kernel<<<(512*1024 + 255)/256, 256, 0, stream>>>(Wih, Whh, bih, bhh,
        Wgbf + (size_t)l*512*1024, bgru + (size_t)l*1024);
  }

  // ---- encoders ----
  atom_encoder_kernel<<<N, 256, 0, stream>>>(x_cat, x_cont,
      (const float*)d_in[6], (const float*)d_in[7], (const float*)d_in[8],
      (const float*)d_in[9], (const float*)d_in[10],
      aW, ab, ag, abe, h, h_bf, N);
  bond_encoder_kernel<<<(E + 3)/4, 256, 0, stream>>>(edge_cat, edge_cont,
      (const float*)d_in[15], (const float*)d_in[16], bW, bb, bg_, bbe, e_bf, E);

  // ---- message passing ----
  for (int l = 0; l < 4; ++l) {
    const float* b1  = (const float*)d_in[21 + l*10 + 1];
    const float* b2  = (const float*)d_in[21 + l*10 + 3];
    const float* lng = (const float*)d_in[21 + l*10 + 8];
    const float* lnb = (const float*)d_in[21 + l*10 + 9];
    hipMemsetAsync(aggr, 0, (size_t)N * HID * 4, stream);
    edge_mlp_kernel<<<(E + 31)/32, 256, 0, stream>>>(h_bf, e_bf, edge_index,
        W1bf + (size_t)l*KIN*HID, b1, W2bf + (size_t)l*HID*HID, b2, aggr, E);
    gru_ln_kernel<<<(N + 15)/16, 256, 0, stream>>>(aggr, h, h_bf,
        Wgbf + (size_t)l*512*1024, bgru + (size_t)l*1024, lng, lnb,
        h, h_bf, N, (l > 0) ? 1 : 0);
  }

  // ---- readout + head ----
  hipMemsetAsync(den, 0, (size_t)G * 4, stream);
  hipMemsetAsync(hgraph, 0, (size_t)G * HID * 4, stream);
  init_gmax_kernel<<<(G + 255)/256, 256, 0, stream>>>(gmax, G);
  gate_kernel<<<N, 128, 0, stream>>>(h, gW1, gb1, gW2, gb2, batch, gates, gmax, N);
  expden_kernel<<<(N + 255)/256, 256, 0, stream>>>(gates, gmax, batch, expg, den, N);
  value_kernel<<<N, 256, 0, stream>>>(h, vW, vb, expg, den, batch, hgraph, N);
  head_kernel<<<G, 128, 0, stream>>>(hgraph, sW, sb, sg, sbe, hW1, hb1, hW2, hb2,
                                     (float*)d_out, G);
}
